// GCN_61040075211230
// MI455X (gfx1250) — matrix operand order repaired and checked by reading, not compiled, not run on hardware
//
#include <hip/hip_runtime.h>
#include <hip/hip_bf16.h>
#include <math.h>

typedef __attribute__((ext_vector_type(16))) _Float16 v16h;
typedef __attribute__((ext_vector_type(8)))  float    v8f;

// ---------------------------------------------------------------------------
// Degree / normalization
// ---------------------------------------------------------------------------
__global__ void gcn_deg_init(float* __restrict__ deg, int n) {
    int i = blockIdx.x * blockDim.x + threadIdx.x;
    if (i < n) deg[i] = 1.0f;                    // +1 self loop
}

__global__ void gcn_deg_edges(const long long* __restrict__ eidx, long long E,
                              float* __restrict__ deg) {
    long long e = (long long)blockIdx.x * blockDim.x + threadIdx.x;
    if (e < E) atomicAdd(&deg[(int)eidx[E + e]], 1.0f);   // dst = eidx[1][e]
}

__global__ void gcn_rsqrt_inplace(float* __restrict__ deg, int n) {
    int i = blockIdx.x * blockDim.x + threadIdx.x;
    if (i < n) deg[i] = rsqrtf(deg[i]);          // becomes dinv
}

// ---------------------------------------------------------------------------
// Dense transform: hW = h @ W   (v_wmma_f32_16x16x32_f16, fp32 accumulate)
// Fused:           agg = hW * dinv^2 + bias    (self-loop + bias term)
// One wave = 16 rows x 32 cols (two 16x16 tiles, K=32, compile-time padded).
// HIN is a template param so fragment builds are branch-free vector loads.
// ---------------------------------------------------------------------------
template <int HIN>
__global__ __launch_bounds__(320)
void gcn_gemm_wmma(const float* __restrict__ hin,
                   const float* __restrict__ W,      // [HIN x 32] row-major
                   const float* __restrict__ bias,   // [32]
                   const float* __restrict__ dinv,   // [N]
                   float* __restrict__ hW,           // [N x 32]
                   float* __restrict__ agg,          // [N x 32]
                   int n)
{
    const int lane   = threadIdx.x & 31;
    const int waveId = blockIdx.x * (blockDim.x >> 5) + (threadIdx.x >> 5);
    const int row0   = waveId << 4;
    if (row0 >= n) return;                       // wave-uniform exit

    const int m   = lane & 15;
    const int hiK = (lane >> 4) << 3;            // 0 for lanes 0-15, 8 for 16-31

    // ---- A fragment (16x32 f16): lane owns row m; element e maps to
    //      K = e + (e&8) + hiK, i.e. two contiguous 8-float runs at hiK, hiK+16.
    v16h a;
    {
        const float* hrow = hin + (size_t)(row0 + m) * HIN;
        const float4 f0 = *reinterpret_cast<const float4*>(hrow + hiK);
        const float4 f1 = *reinterpret_cast<const float4*>(hrow + hiK + 4);
        a[0] = (_Float16)f0.x; a[1] = (_Float16)f0.y;
        a[2] = (_Float16)f0.z; a[3] = (_Float16)f0.w;
        a[4] = (_Float16)f1.x; a[5] = (_Float16)f1.y;
        a[6] = (_Float16)f1.z; a[7] = (_Float16)f1.w;
        if constexpr (HIN == 32) {
            const float4 f2 = *reinterpret_cast<const float4*>(hrow + hiK + 16);
            const float4 f3 = *reinterpret_cast<const float4*>(hrow + hiK + 20);
            a[8]  = (_Float16)f2.x; a[9]  = (_Float16)f2.y;
            a[10] = (_Float16)f2.z; a[11] = (_Float16)f2.w;
            a[12] = (_Float16)f3.x; a[13] = (_Float16)f3.y;
            a[14] = (_Float16)f3.z; a[15] = (_Float16)f3.w;
        } else {                                 // K = 16..31 zero-padded
#pragma unroll
            for (int e = 8; e < 16; ++e) a[e] = (_Float16)0.0f;
        }
    }

    // self-loop coefficient dinv^2 for the 8 output rows this lane owns
    float selfc[8];
#pragma unroll
    for (int r = 0; r < 8; ++r) {
        float dv = dinv[row0 + r + hiK];
        selfc[r] = dv * dv;
    }

    const int kb0 = (lane >> 4) << 3;            // lanes 0-15: 0; 16-31: 8
#pragma unroll
    for (int t = 0; t < 2; ++t) {                // two 16-col tiles
        const int col = (lane & 15) + (t << 4);

        // ---- B fragment (32x16 f16): lane owns column col, K = kb0 + e + (e&8) ----
        v16h bfrag;
        if constexpr (HIN == 32) {
#pragma unroll
            for (int e = 0; e < 16; ++e)
                bfrag[e] = (_Float16)W[(kb0 + e + (e & 8)) * 32 + col];
        } else {                                 // HIN==16: K = 16..31 zero-padded
            {
#pragma unroll
                for (int e = 0; e < 8; ++e)
                    bfrag[e] = (_Float16)W[(kb0 + e) * 32 + col];
#pragma unroll
                for (int e = 8; e < 16; ++e) bfrag[e] = (_Float16)0.0f;
            }
        }

        v8f c = {};
        c = __builtin_amdgcn_wmma_f32_16x16x32_f16(
                /*neg_a=*/false, a, /*neg_b=*/false, bfrag,
                /*c_mod=*/(short)0, c, /*reuse_a=*/false, /*reuse_b=*/false);

        const float bcol = bias[col];
#pragma unroll
        for (int r = 0; r < 8; ++r) {            // D: VGPR r -> row r (+8 hi lanes)
            int outRow = row0 + r + hiK;
            size_t idx = (size_t)outRow * 32 + col;
            float v = c[r];
            hW[idx]  = v;
            agg[idx] = v * selfc[r] + bcol;
        }
    }
}

// ---------------------------------------------------------------------------
// Edge aggregation: agg[dst] += hW[src] * dinv[src]*dinv[dst]
// 8 threads/edge: one b128 gather + 4 fp32 atomics each (L2-resident data).
// ---------------------------------------------------------------------------
__global__ void gcn_edge_agg(const long long* __restrict__ eidx, long long E,
                             const float* __restrict__ dinv,
                             const float* __restrict__ hW,
                             float* __restrict__ agg)
{
    long long t = (long long)blockIdx.x * blockDim.x + threadIdx.x;
    if (t >= E * 8) return;
    long long e  = t >> 3;
    int chunk    = (int)(t & 7);                 // 4 features each
    int s = (int)eidx[e];
    int d = (int)eidx[E + e];
    float coef = dinv[s] * dinv[d];
    const float4 v = *reinterpret_cast<const float4*>(hW + (size_t)s * 32 + chunk * 4);
    float* ap = agg + (size_t)d * 32 + chunk * 4;
    atomicAdd(ap + 0, v.x * coef);
    atomicAdd(ap + 1, v.y * coef);
    atomicAdd(ap + 2, v.z * coef);
    atomicAdd(ap + 3, v.w * coef);
}

__global__ void gcn_tanh(const float* __restrict__ agg, float* __restrict__ h,
                         long long total) {
    long long i = (long long)blockIdx.x * blockDim.x + threadIdx.x;
    if (i < total) h[i] = tanhf(agg[i]);
}

// ---------------------------------------------------------------------------
// Pooling: order-preserving uint flip for float atomicMax
// ---------------------------------------------------------------------------
__device__ __forceinline__ unsigned floatFlip(float f) {
    unsigned u = __float_as_uint(f);
    return (u & 0x80000000u) ? ~u : (u | 0x80000000u);
}
__device__ __forceinline__ float floatUnflip(unsigned u) {
    unsigned r = (u & 0x80000000u) ? (u & 0x7FFFFFFFu) : ~u;
    return __uint_as_float(r);
}

__global__ void pool_init(unsigned* __restrict__ gmaxU, float* __restrict__ gsum,
                          float* __restrict__ cnt, int B) {
    int i = blockIdx.x * blockDim.x + threadIdx.x;
    if (i < B * 32) {
        gmaxU[i] = 0x007FFFFFu;                  // floatFlip(-inf)
        gsum[i]  = 0.0f;
        if (i < B) cnt[i] = 0.0f;
    }
}

__global__ void pool_scatter(const float* __restrict__ h,
                             const long long* __restrict__ batch, int n,
                             unsigned* __restrict__ gmaxU,
                             float* __restrict__ gsum, float* __restrict__ cnt) {
    long long t = (long long)blockIdx.x * blockDim.x + threadIdx.x;
    if (t >= (long long)n * 32) return;
    int node = (int)(t >> 5);
    int f    = (int)(t & 31);
    int b    = (int)batch[node];
    float v  = h[t];
    atomicMax(&gmaxU[b * 32 + f], floatFlip(v));
    atomicAdd(&gsum[b * 32 + f], v);
    if (f == 0) atomicAdd(&cnt[b], 1.0f);
}

// pooled = [max | mean | sum], out = pooled @ W_out + b_out. One block/graph.
__global__ void pool_finalize(const unsigned* __restrict__ gmaxU,
                              const float* __restrict__ gsum,
                              const float* __restrict__ cnt,
                              const float* __restrict__ Wout,
                              const float* __restrict__ bout,
                              float* __restrict__ out,
                              float* __restrict__ pooled) {
    __shared__ float sh[96];
    int b = blockIdx.x;
    int t = threadIdx.x;                         // 0..95
    float v;
    if (t < 32)      v = floatUnflip(gmaxU[b * 32 + t]);
    else if (t < 64) v = gsum[b * 32 + (t - 32)] / fmaxf(cnt[b], 1.0f);
    else             v = gsum[b * 32 + (t - 64)];
    pooled[(size_t)b * 96 + t] = v;
    sh[t] = v * Wout[t];
    __syncthreads();
    if (t == 0) {
        float acc = bout[0];
#pragma unroll
        for (int i = 0; i < 96; ++i) acc += sh[i];
        out[b] = acc;
    }
}

// ---------------------------------------------------------------------------
extern "C" void kernel_launch(void* const* d_in, const int* in_sizes, int n_in,
                              void* d_out, int out_size, void* d_ws, size_t ws_size,
                              hipStream_t stream)
{
    const float*     x     = (const float*)d_in[0];
    const long long* eidx  = (const long long*)d_in[1];
    const long long* batch = (const long long*)d_in[2];
    const float* W0 = (const float*)d_in[3];  const float* b0 = (const float*)d_in[4];
    const float* W1 = (const float*)d_in[5];  const float* b1 = (const float*)d_in[6];
    const float* W2 = (const float*)d_in[7];  const float* b2 = (const float*)d_in[8];
    const float* W3 = (const float*)d_in[9];  const float* b3 = (const float*)d_in[10];
    const float* Wout = (const float*)d_in[11];
    const float* bout = (const float*)d_in[12];

    const int       N = in_sizes[2];                 // batch_index length
    const long long E = (long long)in_sizes[1] / 2;  // edge_index is [2,E]
    const int       B = out_size / 97;               // out(B) + pooled(B*96)

    // workspace layout (fp32), all 16B-aligned
    float* dinv = (float*)d_ws;                      // N
    float* hW   = dinv + N;                          // N*32
    float* agg  = hW + (size_t)N * 32;               // N*32
    float* h0   = agg + (size_t)N * 32;              // N*32
    unsigned* gmaxU = (unsigned*)(h0 + (size_t)N * 32);   // B*32
    float* gsum = (float*)(gmaxU + (size_t)B * 32);  // B*32
    float* cnt  = gsum + (size_t)B * 32;             // B

    // --- normalization coefficients ---
    gcn_deg_init<<<(N + 255) / 256, 256, 0, stream>>>(dinv, N);
    gcn_deg_edges<<<(unsigned)((E + 255) / 256), 256, 0, stream>>>(eidx, E, dinv);
    gcn_rsqrt_inplace<<<(N + 255) / 256, 256, 0, stream>>>(dinv, N);

    const int rowTiles   = (N + 15) / 16;            // 6250
    const int gemmBlocks = (rowTiles + 9) / 10;      // 10 waves / 320-thread block
    const unsigned eBlocks  = (unsigned)((E * 8 + 255) / 256);
    const long long nh      = (long long)N * 32;
    const unsigned nhBlocks = (unsigned)((nh + 255) / 256);

    const float* Ws[4] = {W0, W1, W2, W3};
    const float* bs[4] = {b0, b1, b2, b3};
    for (int layer = 0; layer < 4; ++layer) {
        if (layer == 0) {
            gcn_gemm_wmma<16><<<gemmBlocks, 320, 0, stream>>>(x, Ws[0], bs[0],
                                                              dinv, hW, agg, N);
        } else {
            gcn_gemm_wmma<32><<<gemmBlocks, 320, 0, stream>>>(h0, Ws[layer], bs[layer],
                                                              dinv, hW, agg, N);
        }
        gcn_edge_agg<<<eBlocks, 256, 0, stream>>>(eidx, E, dinv, hW, agg);
        gcn_tanh<<<nhBlocks, 256, 0, stream>>>(agg, h0, nh);
    }

    pool_init<<<(B * 32 + 255) / 256, 256, 0, stream>>>(gmaxU, gsum, cnt, B);
    pool_scatter<<<nhBlocks, 256, 0, stream>>>(h0, batch, N, gmaxU, gsum, cnt);
    pool_finalize<<<B, 96, 0, stream>>>(gmaxU, gsum, cnt, Wout, bout,
                                        (float*)d_out, (float*)d_out + B);
}
